// ObjTokenHungarianLoss_49417893708360
// MI455X (gfx1250) — matrix operand order repaired and checked by reading, not compiled, not run on hardware
//
#include <hip/hip_runtime.h>
#include <math.h>

// ---------------------------------------------------------------------------
// ObjTokenHungarianLoss for MI455X (gfx1250, wave32, WMMA)
//
// Pipeline:
//   K0 init      : zero the 7 stat accumulators in d_ws
//   K1 cost      : per-batch 128x128 cost matrix. Latent pairwise-dot done as
//                  P(128x512) x T^T via v_wmma_f32_16x16x32_f16 with a hi/lo
//                  f16 split (3 WMMAs per tile-kstep => ~fp32 accuracy).
//                  CE / small-chunk terms folded in the epilogue from LDS.
//   K2 lsa       : Jonker-Volgenant assignment, one wave32 per batch.
//                  Columns 4/lane, wave argmin via __shfl_xor, u/p in LDS.
//   K3 matched   : per matched pair compute ce + 6 chunk MSEs, wave-reduce,
//                  atomicAdd into accumulators.
//   K4 finalize  : divide by (B*N + 1e-8), emit [total, stats0..6].
// ---------------------------------------------------------------------------

typedef __attribute__((ext_vector_type(16))) _Float16 v16h;
typedef __attribute__((ext_vector_type(8)))  float    v8f;

#define BN    128
#define DTOT  564
#define NCLS  31
#define LATO  46          // latent offset inside D
#define LATD  512
#define CLIPC 18.420680743952367f   // -log(1e-8)

__device__ __forceinline__ void f32_to_hilo(float x, _Float16& h, _Float16& l) {
  h = (_Float16)x;
  l = (_Float16)(x - (float)h);
}

// -------------------------------- K0 ---------------------------------------
__global__ void init_kernel(float* __restrict__ accum) {
  if (threadIdx.x < 8) accum[threadIdx.x] = 0.f;
}

// -------------------------------- K1 ---------------------------------------
__global__ __launch_bounds__(256) void cost_kernel(const float* __restrict__ pred,
                                                   const float* __restrict__ target,
                                                   float* __restrict__ cost) {
  const int b   = blockIdx.x;
  const int tid = threadIdx.x;

  __shared__ float lse_s[BN];
  __shared__ float pn_s[BN];
  __shared__ float tn_s[BN];
  __shared__ int   cls_s[BN];
  __shared__ float sp_s[BN][18];
  __shared__ float st_s[BN][18];
  __shared__ float lg_s[BN][NCLS];

  // small-feature gather offsets: bmax(3) bmin(3) tr(3) sc(3) rot(6)
  const int OFF[18] = {31,32,33,34,35,36,37,38,39,43,44,45,558,559,560,561,562,563};

  // ---- phase 1: per-row staging (128 pred rows, 128 target rows) ----
  if (tid < BN) {
    const int m = tid;
    const float* row = pred + ((size_t)b*BN + m)*DTOT;
    float mx = row[0];
    #pragma unroll
    for (int k=1;k<NCLS;k++) mx = fmaxf(mx, row[k]);
    float se = 0.f;
    #pragma unroll
    for (int k=0;k<NCLS;k++) { float lg = row[k]; lg_s[m][k] = lg; se += expf(lg-mx); }
    lse_s[m] = mx + logf(se);
    float sn = 0.f;
    #pragma unroll
    for (int i=0;i<18;i++) {
      float p = row[OFF[i]]; sp_s[m][i] = p;
      float w = (i<12) ? (1.f/3.f) : (1.f/6.f);
      sn += w*p*p;
    }
    float ln = 0.f;
    for (int k=0;k<LATD;k++) { float x = row[LATO+k]; ln += x*x; }
    pn_s[m] = ln*(1.f/(float)LATD) + sn;
  } else {
    const int n = tid - BN;
    const float* row = target + ((size_t)b*BN + n)*DTOT;
    float mx = row[0]; int am = 0;
    #pragma unroll
    for (int k=1;k<NCLS;k++) { float x = row[k]; if (x > mx) { mx = x; am = k; } }
    cls_s[n] = am;
    float sn = 0.f;
    #pragma unroll
    for (int i=0;i<18;i++) {
      float t = row[OFF[i]]; st_s[n][i] = t;
      float w = (i<12) ? (1.f/3.f) : (1.f/6.f);
      sn += w*t*t;
    }
    float ln = 0.f;
    for (int k=0;k<LATD;k++) { float x = row[LATO+k]; ln += x*x; }
    tn_s[n] = ln*(1.f/(float)LATD) + sn;
  }
  __syncthreads();

  // ---- phase 2: latent GEMM via WMMA (wave w owns columns 16w..16w+15) ----
  const int wave   = tid >> 5;
  const int lane   = tid & 31;
  const int laneLo = lane & 15;
  const int laneHi = lane >> 4;
  const int n      = wave*16 + laneLo;

  const float* Trow = target + ((size_t)b*BN + n)*DTOT + LATO;

  v8f acc[8];
  {
    v8f z = {0.f,0.f,0.f,0.f,0.f,0.f,0.f,0.f};
    #pragma unroll
    for (int mt=0;mt<8;mt++) acc[mt] = z;
  }

  for (int k=0;k<LATD/32;k++) {
    // B fragment (32x16, B[k][n] = T[n][k]): lane holds 16 contiguous K values
    const int kbB = k*32 + laneHi*16;
    if (k+1 < LATD/32) __builtin_prefetch(Trow + kbB + 32, 0, 0);  // global_prefetch_b8
    v16h bh, bl;
    #pragma unroll
    for (int i=0;i<16;i++) { _Float16 h,l; f32_to_hilo(Trow[kbB+i],h,l); bh[i]=h; bl[i]=l; }

    // A fragment (16x32): lane holds 16 contiguous K values (halves 0..7 at kbA, 8..15 at kbA+8)
    const int kbA = k*32 + laneHi*16;
    #pragma unroll
    for (int mt=0;mt<8;mt++) {
      const int m = mt*16 + laneLo;
      const float* Prow = pred + ((size_t)b*BN + m)*DTOT + LATO;
      v16h ah, al;
      #pragma unroll
      for (int i=0;i<8;i++) { _Float16 h,l; f32_to_hilo(Prow[kbA+i],    h,l); ah[i]  =h; al[i]  =l; }
      #pragma unroll
      for (int i=0;i<8;i++) { _Float16 h,l; f32_to_hilo(Prow[kbA+8+i], h,l); ah[8+i]=h; al[8+i]=l; }

      // hi*hi + hi*lo + lo*hi : near-fp32 dot products with f32 accumulation
      acc[mt] = __builtin_amdgcn_wmma_f32_16x16x32_f16(false, ah, false, bh, (short)0, acc[mt], false, false);
      acc[mt] = __builtin_amdgcn_wmma_f32_16x16x32_f16(false, ah, false, bl, (short)0, acc[mt], false, false);
      acc[mt] = __builtin_amdgcn_wmma_f32_16x16x32_f16(false, al, false, bh, (short)0, acc[mt], false, false);
    }
  }

  // ---- epilogue: combine CE + small-chunk pairwise + latent dot ----
  float stReg[18];
  #pragma unroll
  for (int i=0;i<18;i++) stReg[i] = st_s[n][i];
  const int   cls_n = cls_s[n];
  const float tn_n  = tn_s[n];
  float* costb = cost + (size_t)b*BN*BN;

  #pragma unroll
  for (int mt=0;mt<8;mt++) {
    #pragma unroll
    for (int v=0;v<8;v++) {
      const int m = mt*16 + v + 8*laneHi;         // C layout: VGPR v, lane half
      float cc = fminf(lse_s[m] - lg_s[m][cls_n], CLIPC);
      float cross = 0.f;
      for (int i=0;i<18;i++) {
        float w = (i<12) ? (1.f/3.f) : (1.f/6.f);
        cross += w * sp_s[m][i]*stReg[i];
      }
      costb[(size_t)m*BN + n] =
          cc + pn_s[m] + tn_n - 2.f*cross - (2.f/(float)LATD)*acc[mt][v];
    }
  }
}

// -------------------------------- K2 ---------------------------------------
// Jonker-Volgenant, one wave32 per batch. Column j (1-based) owned by
// lane (j-1)&31, slot (j-1)>>5. u[], p[] in LDS; v, minv, way, used in regs.
__global__ __launch_bounds__(32) void lsa_kernel(const float* __restrict__ cost,
                                                 int* __restrict__ rows) {
  const int b    = blockIdx.x;
  const int lane = threadIdx.x;
  __shared__ float u_s[BN+1];
  __shared__ int   p_s[BN+1];
  const float* C = cost + (size_t)b*BN*BN;

  for (int j = lane; j <= BN; j += 32) { u_s[j] = 0.f; p_s[j] = 0; }
  float v_r[4];
  for (int s=0;s<4;s++) v_r[s] = 0.f;
  __syncthreads();

  for (int i=1;i<=BN;i++) {
    if (lane==0) p_s[0] = i;
    float minv[4]; int way[4]; int used[4];
    #pragma unroll
    for (int s=0;s<4;s++) { minv[s] = __builtin_inff(); way[s]=0; used[s]=0; }
    __syncthreads();

    int j0 = 0;
    while (true) {
      if (j0 != 0) {                                  // used[j0] = true
        int sl = (j0-1)>>5, ow = (j0-1)&31;
        if (lane==ow) used[sl] = 1;
      }
      const int i0 = p_s[j0];
      const float ui0 = u_s[i0];
      const float* crow = C + (size_t)(i0-1)*BN;

      float bestv = __builtin_inff(); int bestj = 0x7fffffff;
      #pragma unroll
      for (int s=0;s<4;s++) {
        if (!used[s]) {
          const int j = lane + 1 + 32*s;
          float cur = crow[j-1] - ui0 - v_r[s];
          if (cur < minv[s]) { minv[s] = cur; way[s] = j0; }
          if (minv[s] < bestv || (minv[s]==bestv && j < bestj)) { bestv = minv[s]; bestj = j; }
        }
      }
      // wave argmin, tie -> smallest j (matches np.argmin first-hit)
      #pragma unroll
      for (int off=16; off>0; off>>=1) {
        float ov = __shfl_xor(bestv, off);
        int   oj = __shfl_xor(bestj, off);
        if (ov < bestv || (ov==bestv && oj < bestj)) { bestv = ov; bestj = oj; }
      }
      const float delta = bestv;
      const int   j1    = bestj;

      if (lane==0) u_s[p_s[0]] += delta;              // j=0 always "used"
      #pragma unroll
      for (int s=0;s<4;s++) {
        if (used[s]) {
          const int j = lane + 1 + 32*s;
          u_s[p_s[j]] += delta;                       // p values distinct -> race-free
          v_r[s] -= delta;
        } else {
          minv[s] -= delta;
        }
      }
      __syncthreads();
      j0 = j1;
      if (p_s[j0] == 0) break;
    }

    // augment along `way` chain
    while (j0 != 0) {
      const int sl = (j0-1)>>5, ow = (j0-1)&31;
      int cand = way[sl];
      const int j1 = __shfl(cand, ow);
      const int pj1 = p_s[j1];
      __syncthreads();
      if (lane==0) p_s[j0] = pj1;
      __syncthreads();
      j0 = j1;
    }
  }
  __syncthreads();
  for (int j = lane+1; j <= BN; j += 32) rows[b*BN + (j-1)] = p_s[j] - 1;
}

// -------------------------------- K3 ---------------------------------------
__global__ __launch_bounds__(128) void matched_kernel(const float* __restrict__ pred,
                                                      const float* __restrict__ target,
                                                      const int* __restrict__ rows,
                                                      float* __restrict__ accum) {
  const int b = blockIdx.x;
  const int t = threadIdx.x;
  const int r = rows[b*BN + t];
  const float* P = pred   + ((size_t)b*BN + r)*DTOT;
  const float* T = target + ((size_t)b*BN + t)*DTOT;

  // gt = argmax(target logits)
  float mx = T[0]; int gt = 0;
  #pragma unroll
  for (int k=1;k<NCLS;k++) { float x = T[k]; if (x > mx) { mx = x; gt = k; } }
  // ce = logsumexp(pred logits) - logit[gt]
  float pmx = P[0];
  #pragma unroll
  for (int k=1;k<NCLS;k++) pmx = fmaxf(pmx, P[k]);
  float se = 0.f;
  #pragma unroll
  for (int k=0;k<NCLS;k++) se += expf(P[k]-pmx);
  float vals[7];
  vals[0] = (pmx + logf(se)) - P[gt];

  float a;
  a=0.f; for (int i=0;i<3;i++){ float d=P[31+i] -T[31+i];  a+=d*d; } vals[1]=a*(1.f/3.f); // bmax
  a=0.f; for (int i=0;i<3;i++){ float d=P[34+i] -T[34+i];  a+=d*d; } vals[2]=a*(1.f/3.f); // bmin
  a=0.f; for (int i=0;i<3;i++){ float d=P[37+i] -T[37+i];  a+=d*d; } vals[3]=a*(1.f/3.f); // tr
  a=0.f; for (int i=0;i<6;i++){ float d=P[558+i]-T[558+i]; a+=d*d; } vals[4]=a*(1.f/6.f); // rot
  a=0.f; for (int i=0;i<3;i++){ float d=P[43+i] -T[43+i];  a+=d*d; } vals[5]=a*(1.f/3.f); // sc
  a=0.f; for (int k=0;k<LATD;k++){ float d=P[LATO+k]-T[LATO+k]; a+=d*d; }
  vals[6]=a*(1.f/(float)LATD);                                                            // la

  #pragma unroll
  for (int off=16; off>0; off>>=1) {
    #pragma unroll
    for (int k=0;k<7;k++) vals[k] += __shfl_down(vals[k], off);
  }
  if ((threadIdx.x & 31) == 0) {
    #pragma unroll
    for (int k=0;k<7;k++) atomicAdd(&accum[k], vals[k]);
  }
}

// -------------------------------- K4 ---------------------------------------
__global__ void finalize_kernel(const float* __restrict__ accum,
                                float* __restrict__ out, int count) {
  if (threadIdx.x == 0 && blockIdx.x == 0) {
    const double denom = (double)count + 1e-8;
    double tot = 0.0;
    for (int k=0;k<7;k++) tot += (double)accum[k];
    out[0] = (float)(tot / denom);
    for (int k=0;k<7;k++) out[1+k] = (float)((double)accum[k] / denom);
  }
}

// ---------------------------------------------------------------------------
extern "C" void kernel_launch(void* const* d_in, const int* in_sizes, int n_in,
                              void* d_out, int out_size, void* d_ws, size_t ws_size,
                              hipStream_t stream) {
  const float* pred   = (const float*)d_in[0];
  const float* target = (const float*)d_in[1];
  // d_in[2] (attention_mask) is unused by the reference.

  const int B = in_sizes[0] / (BN * DTOT);   // 256 for the reference setup

  const size_t cost_bytes = (size_t)B * BN * BN * sizeof(float);
  const size_t rows_bytes = (size_t)B * BN * sizeof(int);
  float* cost  = (float*)d_ws;
  int*   rows  = (int*)((char*)d_ws + cost_bytes);
  float* accum = (float*)((char*)d_ws + cost_bytes + rows_bytes);

  init_kernel    <<<1,   32,  0, stream>>>(accum);
  cost_kernel    <<<B,   256, 0, stream>>>(pred, target, cost);
  lsa_kernel     <<<B,   32,  0, stream>>>(cost, rows);
  matched_kernel <<<B,   128, 0, stream>>>(pred, target, rows, accum);
  finalize_kernel<<<1,   32,  0, stream>>>(accum, (float*)d_out, B*BN);
}
